// GAT_60627758350589
// MI455X (gfx1250) — matrix operand order repaired and checked by reading, not compiled, not run on hardware
//
#include <hip/hip_runtime.h>
#include <hip/hip_bf16.h>

typedef __attribute__((ext_vector_type(16))) _Float16 v16h;
typedef __attribute__((ext_vector_type(8)))  float    v8f;

#define LRELU(x, s) ((x) > 0.f ? (x) : (s) * (x))

__device__ __forceinline__ float wave_sum(float v) {
#pragma unroll
  for (int o = 16; o > 0; o >>= 1) v += __shfl_xor(v, o, 32);
  return v;
}
// order-preserving f32 -> u32 map for atomicMax on floats
__device__ __forceinline__ unsigned fenc(float x) {
  unsigned u = __float_as_uint(x);
  return (u >> 31) ? ~u : (u | 0x80000000u);
}
__device__ __forceinline__ float fdec(unsigned u) {
  return __uint_as_float((u >> 31) ? (u & 0x7fffffffu) : ~u);
}

// ---------------------------------------------------------------------------
// Weight repack: W[K,NCOLS] f32 -> f16 fragments in this GEMM's B layout (k contiguous per lane half).
// v16h fragment index = (tile*(K/32) + kstep)*32 + lane; halfs [2j,2j+1] are
// W[k][n], W[k+1][n] with k = kstep*32 + (lane&16 ? 16:0) + 2j, n = tile*16+(lane&15).
// ---------------------------------------------------------------------------
__global__ void repack_w(const float* __restrict__ W, _Float16* __restrict__ PB,
                         int K, int NCOLS) {
  int p = blockIdx.x * blockDim.x + threadIdx.x;      // pair index
  int total = K * NCOLS / 2;
  if (p >= total) return;
  int j     = p & 7;
  int lane  = (p >> 3) & 31;
  int rest  = p >> 8;
  int ksteps = K / 32;
  int kstep = rest % ksteps;
  int tile  = rest / ksteps;
  int k = kstep * 32 + ((lane & 16) ? 16 : 0) + 2 * j;
  int n = tile * 16 + (lane & 15);
  PB[2 * p]     = (_Float16)W[(size_t)k * NCOLS + n];
  PB[2 * p + 1] = (_Float16)W[(size_t)(k + 1) * NCOLS + n];
}

// ---------------------------------------------------------------------------
// WMMA GEMM: C[M,NCOLS] = op(A)[M,K] @ W[K,NCOLS] (+bias). op = LayerNorm if LN.
// Block = (NCOLS/16) waves; 64-row M-tile per block (MT=4 16-row subtiles).
// A staged in LDS as f16 (padded stride); B from pre-packed f16 fragments
// (one 32B vector load per K-step, reused across MT subtiles).
// ---------------------------------------------------------------------------
template <int K, int NCOLS, int MT, bool LN>
__global__ __launch_bounds__(256) void gemm_wmma(
    const float* __restrict__ A, const _Float16* __restrict__ PB,
    const float* __restrict__ bias,
    const float* __restrict__ lng, const float* __restrict__ lnb,
    float* __restrict__ C, int M) {
  constexpr int KP   = K + 8;              // bank-spread padding
  constexpr int ROWS = 16 * MT;
  __shared__ _Float16 sA[ROWS * KP];
  const int tid  = threadIdx.x;
  const int lane = tid & 31;
  const int wave = tid >> 5;
  const int m0   = blockIdx.x * ROWS;

  if (LN) {                                // 8 waves x (ROWS/8) rows
#pragma unroll
    for (int r2 = 0; r2 < ROWS / 8; ++r2) {
      int row  = wave * (ROWS / 8) + r2;
      int grow = m0 + row;
      if (grow < M) {
        float v[K / 32];
        float s = 0.f;
#pragma unroll
        for (int j = 0; j < K / 32; ++j) {
          v[j] = A[(size_t)grow * K + lane + 32 * j];
          s += v[j];
        }
        s = wave_sum(s);
        float mean = s * (1.0f / K);
        float q = 0.f;
#pragma unroll
        for (int j = 0; j < K / 32; ++j) { float d = v[j] - mean; q += d * d; }
        q = wave_sum(q);
        float rstd = rsqrtf(q * (1.0f / K) + 1e-5f);
#pragma unroll
        for (int j = 0; j < K / 32; ++j) {
          int f = lane + 32 * j;
          sA[row * KP + f] = (_Float16)((v[j] - mean) * rstd * lng[f] + lnb[f]);
        }
      } else {
#pragma unroll
        for (int j = 0; j < K / 32; ++j) sA[row * KP + lane + 32 * j] = (_Float16)0.f;
      }
    }
  } else {
    for (int i = tid; i < ROWS * K; i += blockDim.x) {
      int row = i / K, col = i - row * K;
      int grow = m0 + row;
      sA[row * KP + col] = (_Float16)(grow < M ? A[(size_t)grow * K + col] : 0.f);
    }
  }
  __syncthreads();

  const int nloc = wave * 16 + (lane & 15);
  const int mrow = lane & 15;
  const int aoff = (lane & 16) ? 16 : 0;   // A: lanes 16-31 hold K half +16
  const v16h* __restrict__ PBv = (const v16h*)PB;
  int bidx = wave * (K / 32) * 32 + lane;

  v8f acc[MT];
#pragma unroll
  for (int t = 0; t < MT; ++t) acc[t] = (v8f){};

  for (int kk = 0; kk < K; kk += 32) {
    v16h bf = PBv[bidx];
    bidx += 32;
#pragma unroll
    for (int t = 0; t < MT; ++t) {
      union { v16h h; unsigned u[8]; } af;
      const unsigned* p0 = (const unsigned*)&sA[(mrow + 16 * t) * KP + kk + aoff];
      const unsigned* p1 = (const unsigned*)&sA[(mrow + 16 * t) * KP + kk + 8 + aoff];
#pragma unroll
      for (int j = 0; j < 4; ++j) { af.u[j] = p0[j]; af.u[4 + j] = p1[j]; }
      acc[t] = __builtin_amdgcn_wmma_f32_16x16x32_f16(false, af.h, false, bf,
                                                      (short)0, acc[t], false, false);
    }
  }
#pragma unroll
  for (int t = 0; t < MT; ++t) {
#pragma unroll
    for (int r = 0; r < 8; ++r) {
      int mm = (lane & 16) ? (8 + r) : r;  // C layout: VGPR r = row r / 8+r
      int grow = m0 + 16 * t + mm;
      if (grow < M) {
        float val = acc[t][r];
        if (bias) val += bias[nloc];
        C[(size_t)grow * NCOLS + nloc] = val;
      }
    }
  }
}

// ---------------------------------------------------------------------------
// Small helpers / edge-wise kernels
// ---------------------------------------------------------------------------
__global__ void zero2_kernel(float* a, float* b, int n) {
  int i = blockIdx.x * blockDim.x + threadIdx.x;
  if (i < n) { a[i] = 0.f; b[i] = 0.f; }
}

__global__ void degree_kernel(const int* __restrict__ dst, const float* __restrict__ ea,
                              float* cnt, float* esum, int E_) {
  int e = blockIdx.x * blockDim.x + threadIdx.x;
  if (e >= E_) return;
  int d = dst[e];
  atomicAdd(&cnt[d], 1.f);
  atomicAdd(&esum[d], ea[e]);
}

__global__ void dot_small(const float* __restrict__ a, const float* __restrict__ b,
                          int F, float* out) {
  __shared__ float red[128];
  int t = threadIdx.x;
  float p = 0.f;
  for (int i = t; i < F; i += 128) p += a[i] * b[i];
  red[t] = p;
  __syncthreads();
  for (int s = 64; s > 0; s >>= 1) {
    if (t < s) red[t] += red[t + s];
    __syncthreads();
  }
  if (t == 0) out[0] = red[0];
}

// per node: als/ald dots, self-loop logit, init running max
__global__ void node_attn(const float* __restrict__ t, const float* __restrict__ a_s,
                          const float* __restrict__ a_d, const float* __restrict__ cnt,
                          const float* __restrict__ esum, const float* __restrict__ ce,
                          float* als, float* ald, float* sl, unsigned* mm, int F, int n) {
  int gid = blockIdx.x * blockDim.x + threadIdx.x;
  int node = gid >> 5, lane = gid & 31;
  if (node >= n) return;
  float ps = 0.f, pd = 0.f;
  for (int f = lane; f < F; f += 32) {
    float tv = t[(size_t)node * F + f];
    ps += tv * a_s[f];
    pd += tv * a_d[f];
  }
  ps = wave_sum(ps);
  pd = wave_sum(pd);
  if (lane == 0) {
    float le = esum[node] / fmaxf(cnt[node], 1.f) * ce[0];
    float lg = LRELU(ps + pd + le, 0.2f);
    als[node] = ps; ald[node] = pd; sl[node] = lg; mm[node] = fenc(lg);
  }
}

__global__ void edge_logit(const int* __restrict__ src, const int* __restrict__ dst,
                           const float* __restrict__ ea, const float* __restrict__ als,
                           const float* __restrict__ ald, const float* __restrict__ ce,
                           float* elog, unsigned* mm, int E_) {
  int e = blockIdx.x * blockDim.x + threadIdx.x;
  if (e >= E_) return;
  int s = src[e], d = dst[e];
  float lg = LRELU(als[s] + ald[d] + ea[e] * ce[0], 0.2f);
  elog[e] = lg;
  atomicMax(&mm[d], fenc(lg));
}

// decode max in place, init denominator with self-loop term
__global__ void node_den(const unsigned* mmIn, float* mfOut, const float* sl,
                         float* den, int n) {
  int i = blockIdx.x * blockDim.x + threadIdx.x;
  if (i >= n) return;
  float m = fdec(mmIn[i]);
  mfOut[i] = m;                       // aliases mmIn (same slot, same thread)
  den[i] = __expf(sl[i] - m);
}

__global__ void edge_den(const int* __restrict__ dst, const float* __restrict__ elog,
                         const float* __restrict__ mf, float* den, int E_) {
  int e = blockIdx.x * blockDim.x + threadIdx.x;
  if (e >= E_) return;
  int d = dst[e];
  atomicAdd(&den[d], __expf(elog[e] - mf[d]));
}

template <int F>
__global__ void self_agg(const float* __restrict__ t, const float* __restrict__ sl,
                         const float* __restrict__ mf, const float* __restrict__ den,
                         float* agg, int total) {
  int i = blockIdx.x * blockDim.x + threadIdx.x;
  if (i >= total) return;
  int node = i / F;
  float wgt = __expf(sl[node] - mf[node]) / den[node];
  agg[i] = wgt * t[i];
}

// one wave per edge; lane owns F/32 contiguous features
template <int F>
__global__ void edge_agg(const int* __restrict__ src, const int* __restrict__ dst,
                         const float* __restrict__ elog, const float* __restrict__ mf,
                         const float* __restrict__ den, const float* __restrict__ t,
                         float* agg, int E_) {
  int gid = blockIdx.x * blockDim.x + threadIdx.x;
  int e = gid >> 5, lane = gid & 31;
  if (e >= E_) return;
  int s = src[e], d = dst[e];
  float wgt = __expf(elog[e] - mf[d]) / den[d];
  constexpr int CPL = F / 32;
  const float* tp = t + (size_t)s * F + lane * CPL;
  float* ap = agg + (size_t)d * F + lane * CPL;
  float v[CPL];
#pragma unroll
  for (int i = 0; i < CPL; ++i) v[i] = tp[i];
#pragma unroll
  for (int i = 0; i < CPL; ++i) atomicAdd(ap + i, v[i] * wgt);
}

template <int F>
__global__ void bias_act(float* h, const float* __restrict__ bias, int total) {
  int i = blockIdx.x * blockDim.x + threadIdx.x;
  if (i >= total) return;
  float v = h[i] + bias[i & (F - 1)];
  h[i] = LRELU(v, 0.01f);
}

// fused LayerNorm(64) + fc2(64 -> 2); one wave per node
__global__ void final_kernel(const float* __restrict__ h, const float* __restrict__ g,
                             const float* __restrict__ bb, const float* __restrict__ fw,
                             const float* __restrict__ fb, float* __restrict__ out, int n) {
  int gid = blockIdx.x * blockDim.x + threadIdx.x;
  int node = gid >> 5, lane = gid & 31;
  if (node >= n) return;
  float v0 = h[(size_t)node * 64 + lane];
  float v1 = h[(size_t)node * 64 + 32 + lane];
  float s = wave_sum(v0 + v1);
  float mean = s * (1.f / 64.f);
  float d0 = v0 - mean, d1 = v1 - mean;
  float q = wave_sum(d0 * d0 + d1 * d1);
  float rstd = rsqrtf(q * (1.f / 64.f) + 1e-5f);
  float x0 = d0 * rstd * g[lane] + bb[lane];
  float x1 = d1 * rstd * g[lane + 32] + bb[lane + 32];
  float p0 = x0 * fw[lane * 2]     + x1 * fw[(lane + 32) * 2];
  float p1 = x0 * fw[lane * 2 + 1] + x1 * fw[(lane + 32) * 2 + 1];
  p0 = wave_sum(p0);
  p1 = wave_sum(p1);
  if (lane == 0) {
    out[node * 2]     = p0 + fb[0];
    out[node * 2 + 1] = p1 + fb[1];
  }
}

// ---------------------------------------------------------------------------
extern "C" void kernel_launch(void* const* d_in, const int* in_sizes, int n_in,
                              void* d_out, int out_size, void* d_ws, size_t ws_size,
                              hipStream_t stream) {
  const float* x     = (const float*)d_in[0];
  const int*   eidx  = (const int*)  d_in[1];
  const float* ea    = (const float*)d_in[2];
  const float* ln1_g = (const float*)d_in[3];
  const float* ln1_b = (const float*)d_in[4];
  const float* fc1_w = (const float*)d_in[5];
  const float* fc1_b = (const float*)d_in[6];
  const float* c1_w  = (const float*)d_in[7];
  const float* c1_as = (const float*)d_in[8];
  const float* c1_ad = (const float*)d_in[9];
  const float* c1_we = (const float*)d_in[10];
  const float* c1_ae = (const float*)d_in[11];
  const float* c1_b  = (const float*)d_in[12];
  const float* c2_w  = (const float*)d_in[13];
  const float* c2_as = (const float*)d_in[14];
  const float* c2_ad = (const float*)d_in[15];
  const float* c2_we = (const float*)d_in[16];
  const float* c2_ae = (const float*)d_in[17];
  const float* c2_b  = (const float*)d_in[18];
  const float* ln2_g = (const float*)d_in[19];
  const float* ln2_b = (const float*)d_in[20];
  const float* fc2_w = (const float*)d_in[21];
  const float* fc2_b = (const float*)d_in[22];

  const int N = in_sizes[0] / 384;
  const int E = in_sizes[1] / 2;
  const int* src = eidx;
  const int* dst = eidx + E;

  float* w = (float*)d_ws;
  float* bufA = w; w += (size_t)N * 128;   // fc1 out / conv2 features
  float* bufB = w; w += (size_t)N * 128;   // conv1 features / conv2 agg
  float* bufC = w; w += (size_t)N * 128;   // conv1 agg -> conv2 input
  float* elog = w; w += E;
  float* als  = w; w += N;
  float* ald  = w; w += N;
  float* sl   = w; w += N;
  float* mf   = w; w += N;                 // u32 max, decoded in place to f32
  float* den  = w; w += N;
  float* cnt  = w; w += N;
  float* esum = w; w += N;
  float* ce   = w; w += 8;                 // scalar + pad to keep 32B alignment
  _Float16* pb1 = (_Float16*)w; w += (384 * 128) / 2;   // packed fc1_w
  _Float16* pb2 = (_Float16*)w; w += (128 * 128) / 2;   // packed conv1_w
  _Float16* pb3 = (_Float16*)w; w += (128 * 64) / 2;    // packed conv2_w
  unsigned* mU = (unsigned*)mf;

  const int gN64   = (N + 63) / 64;
  const int gNwave = (N + 7) / 8;
  const int gN256  = (N + 255) / 256;
  const int gE     = (E + 255) / 256;
  const int gEwave = (E + 7) / 8;

  // one-time (per launch) weight repacks into WMMA B-fragment layout
  repack_w<<<(384 * 128 / 2 + 255) / 256, 256, 0, stream>>>(fc1_w, pb1, 384, 128);
  repack_w<<<(128 * 128 / 2 + 255) / 256, 256, 0, stream>>>(c1_w, pb2, 128, 128);
  repack_w<<<(128 * 64 / 2 + 255) / 256, 256, 0, stream>>>(c2_w, pb3, 128, 64);

  zero2_kernel<<<gN256, 256, 0, stream>>>(cnt, esum, N);
  degree_kernel<<<gE, 256, 0, stream>>>(dst, ea, cnt, esum, E);

  // LN(x) @ fc1_w + fc1_b
  gemm_wmma<384, 128, 4, true><<<gN64, 256, 0, stream>>>(x, pb1, fc1_b, ln1_g, ln1_b, bufA, N);

  // ---- GAT layer 1 (F = 128) ----
  dot_small<<<1, 128, 0, stream>>>(c1_we, c1_ae, 128, ce);
  gemm_wmma<128, 128, 4, false><<<gN64, 256, 0, stream>>>(bufA, pb2, nullptr, nullptr, nullptr, bufB, N);
  node_attn<<<gNwave, 256, 0, stream>>>(bufB, c1_as, c1_ad, cnt, esum, ce, als, ald, sl, mU, 128, N);
  edge_logit<<<gE, 256, 0, stream>>>(src, dst, ea, als, ald, ce, elog, mU, E);
  node_den<<<gN256, 256, 0, stream>>>(mU, mf, sl, den, N);
  edge_den<<<gE, 256, 0, stream>>>(dst, elog, mf, den, E);
  self_agg<128><<<(N * 128 + 255) / 256, 256, 0, stream>>>(bufB, sl, mf, den, bufC, N * 128);
  edge_agg<128><<<gEwave, 256, 0, stream>>>(src, dst, elog, mf, den, bufB, bufC, E);
  bias_act<128><<<(N * 128 + 255) / 256, 256, 0, stream>>>(bufC, c1_b, N * 128);

  // ---- GAT layer 2 (F = 64) ----
  dot_small<<<1, 128, 0, stream>>>(c2_we, c2_ae, 64, ce);
  gemm_wmma<128, 64, 4, false><<<gN64, 128, 0, stream>>>(bufC, pb3, nullptr, nullptr, nullptr, bufA, N);
  node_attn<<<gNwave, 256, 0, stream>>>(bufA, c2_as, c2_ad, cnt, esum, ce, als, ald, sl, mU, 64, N);
  edge_logit<<<gE, 256, 0, stream>>>(src, dst, ea, als, ald, ce, elog, mU, E);
  node_den<<<gN256, 256, 0, stream>>>(mU, mf, sl, den, N);
  edge_den<<<gE, 256, 0, stream>>>(dst, elog, mf, den, E);
  self_agg<64><<<(N * 64 + 255) / 256, 256, 0, stream>>>(bufA, sl, mf, den, bufB, N * 64);
  edge_agg<64><<<gEwave, 256, 0, stream>>>(src, dst, elog, mf, den, bufA, bufB, E);
  bias_act<64><<<(N * 64 + 255) / 256, 256, 0, stream>>>(bufB, c2_b, N * 64);

  // LN(h) @ fc2_w + fc2_b
  final_kernel<<<gNwave, 256, 0, stream>>>(bufB, ln2_g, ln2_b, fc2_w, fc2_b, (float*)d_out, N);

  (void)n_in; (void)out_size; (void)ws_size;
}
